// MultiHeadAttention_7593502179368
// MI455X (gfx1250) — matrix operand order repaired and checked by reading, not compiled, not run on hardware
//
#include <hip/hip_runtime.h>
#include <hip/hip_bf16.h>

#define D_MODEL 1024
#define SEQ     2048
#define NH      16
#define DK      64
#define BATCH   2
#define SROW    2056   // LDS f32 score row stride (floats), padded vs 2048
#define PROW    2064   // LDS bf16 probability row stride (ushorts), padded

typedef __attribute__((ext_vector_type(16))) __bf16        v16bf;
typedef __attribute__((ext_vector_type(8)))  float         v8f;
typedef __attribute__((ext_vector_type(8)))  unsigned int  v8u;

// ---- fp32 -> bf16 (round-to-nearest-even) helpers (only used OUTSIDE hot loops) ----
__device__ __forceinline__ unsigned int pk_bf16(float a, float b) {
    unsigned int ua = __float_as_uint(a), ub = __float_as_uint(b);
    ua = (ua + 0x7FFFu + ((ua >> 16) & 1u)) >> 16;
    ub = (ub + 0x7FFFu + ((ub >> 16) & 1u)) >> 16;
    return (ua & 0xFFFFu) | (ub << 16);
}
__device__ __forceinline__ unsigned short f2bf(float a) {
    unsigned int ua = __float_as_uint(a);
    return (unsigned short)((ua + 0x7FFFu + ((ua >> 16) & 1u)) >> 16);
}

// ---- WMMA fragment loaders (this code's contiguous k order, k = 16*(L/16) + s, shared by A and B) ----
// A 16x32 bf16: lane L -> row M = L&15; lanes 0-15 hold K0..15, lanes 16-31 hold K16..31
__device__ __forceinline__ v16bf load_A_b16(const unsigned short* base, int ld, int mbase,
                                            int k0, int lane) {
    const unsigned int* p = (const unsigned int*)(base +
        (size_t)(mbase + (lane & 15)) * ld + k0 + ((lane >> 4) << 4));
    v8u u;
#pragma unroll
    for (int i = 0; i < 4; ++i) u[i]     = p[i];
#pragma unroll
    for (int i = 0; i < 4; ++i) u[4 + i] = p[4 + i];
    return __builtin_bit_cast(v16bf, u);
}
// B 32x16 bf16: lane L -> col N = L&15; lanes 0-15 hold K0..15, lanes 16-31 hold K16..31.
// Column N is contiguous at base + N*ld (weights row-major; V stored [dk,S]).
__device__ __forceinline__ v16bf load_B_b16(const unsigned short* base, int ld, int nbase,
                                            int k0, int lane) {
    const unsigned int* p = (const unsigned int*)(base +
        (size_t)(nbase + (lane & 15)) * ld + k0 + ((lane >> 4) << 4));
    v8u u;
#pragma unroll
    for (int i = 0; i < 8; ++i) u[i] = p[i];
    return __builtin_bit_cast(v16bf, u);
}

#define WMMA_BF16(a, b, c) \
    __builtin_amdgcn_wmma_f32_16x16x32_bf16(false, (a), false, (b), (short)0, (c), false, false)

// =====================================================================
// Kernel 0: fp32 -> bf16 bulk conversion (bandwidth-bound, one pass).
// =====================================================================
__global__ void __launch_bounds__(256)
mha_cvt_bf16(const float* __restrict__ src, unsigned short* __restrict__ dst, int n) {
    int base = (blockIdx.x * 256 + threadIdx.x) * 8;
    if (base >= n) return;
    const float4* s = (const float4*)(src + base);
    float4 f0 = s[0], f1 = s[1];
    uint4 o;
    o.x = pk_bf16(f0.x, f0.y);
    o.y = pk_bf16(f0.z, f0.w);
    o.z = pk_bf16(f1.x, f1.y);
    o.w = pk_bf16(f1.z, f1.w);
    *(uint4*)(dst + base) = o;
}

// =====================================================================
// Kernel 1: fused Q/K/V projections, bf16 in/out.  y = x @ W.T + b.
// Each wave owns a 64x32 output block (4x2 WMMA tiles): 6 fragment loads
// (12 b128) feed 8 v_wmma per K-step -> 1.5 loads/WMMA.
// Q,K -> [B,H,S,dk] (Q pre-scaled by 1/sqrt(dk));  V -> [B,H,dk,S].
// =====================================================================
__global__ void __launch_bounds__(256)
mha_qkv_proj(const unsigned short* __restrict__ xb,
             const unsigned short* __restrict__ Wqb,
             const unsigned short* __restrict__ Wkb,
             const unsigned short* __restrict__ Wvb,
             const float* __restrict__ bq, const float* __restrict__ bk,
             const float* __restrict__ bv,
             unsigned short* __restrict__ Qw, unsigned short* __restrict__ Kw,
             unsigned short* __restrict__ Vt) {
    const int wid   = blockIdx.x * 8 + (threadIdx.x >> 5);
    const int lane  = threadIdx.x & 31;
    const int which = wid >> 11;             // /2048 : 0=Q 1=K 2=V
    const int rem   = wid & 2047;
    const int m0    = (rem >> 5) << 6;       // 64 M-blocks of 64 rows over B*S=4096
    const int n0    = (rem & 31) << 5;       // 32 N-blocks of 32 cols over 1024

    const unsigned short* W;
    const float* bvec;
    if (which == 0)      { W = Wqb; bvec = bq; }
    else if (which == 1) { W = Wkb; bvec = bk; }
    else                 { W = Wvb; bvec = bv; }

    const unsigned short* aRow = xb + (size_t)(m0 + (lane & 15)) * D_MODEL;
    const unsigned short* bRow = W  + (size_t)(n0 + (lane & 15)) * D_MODEL;

    v8f acc[4][2];
#pragma unroll
    for (int tm = 0; tm < 4; ++tm)
#pragma unroll
        for (int tn = 0; tn < 2; ++tn) acc[tm][tn] = (v8f){};

    for (int k = 0; k < D_MODEL; k += 32) {
        if (k + 32 < D_MODEL) {
            __builtin_prefetch(aRow + k + 32, 0, 1);
            __builtin_prefetch(bRow + k + 32, 0, 1);
        }
        v16bf a[4], b[2];
#pragma unroll
        for (int tm = 0; tm < 4; ++tm)
            a[tm] = load_A_b16(xb, D_MODEL, m0 + tm * 16, k, lane);
#pragma unroll
        for (int tn = 0; tn < 2; ++tn)
            b[tn] = load_B_b16(W, D_MODEL, n0 + tn * 16, k, lane);
#pragma unroll
        for (int tm = 0; tm < 4; ++tm)
#pragma unroll
            for (int tn = 0; tn < 2; ++tn)
                acc[tm][tn] = WMMA_BF16(a[tm], b[tn], acc[tm][tn]);
    }

#pragma unroll
    for (int tm = 0; tm < 4; ++tm) {
#pragma unroll
        for (int tn = 0; tn < 2; ++tn) {
            const int   n    = n0 + tn * 16 + (lane & 15);
            const float bias = bvec[n];
            const int   h    = n >> 6, d = n & (DK - 1);
#pragma unroll
            for (int i = 0; i < 8; ++i) {
                int   m  = m0 + tm * 16 + ((lane >> 4) << 3) + i;
                int   bb = m >> 11, s = m & (SEQ - 1);
                float v  = acc[tm][tn][i] + bias;
                if (which == 0) v *= 0.125f;   // 1/sqrt(64)
                if (which == 2)
                    Vt[(((size_t)bb * NH + h) * DK + d) * SEQ + s] = f2bf(v);
                else {
                    unsigned short* dst = (which == 0) ? Qw : Kw;
                    dst[(((size_t)bb * NH + h) * SEQ + s) * DK + d] = f2bf(v);
                }
            }
        }
    }
}

// =====================================================================
// Kernel 2: causal attention. One block per (b, h, 16-query-row block).
// Pass 1 (8 waves): WMMA score strip into LDS (causal tiles only).
// Softmax: f32 reductions, probabilities written ONCE to LDS as bf16.
// Pass 2 (8 waves = 2 n-tile-pairs x 4 K-quarters): P fragment reused
// across two V tiles; partial f32 tiles in disjoint LDS slots, summed
// + scaled by 1/rowsum at writeout.
// =====================================================================
__global__ void __launch_bounds__(256)
mha_attention(const unsigned short* __restrict__ Qw,
              const unsigned short* __restrict__ Kw,
              const unsigned short* __restrict__ Vt,
              unsigned short* __restrict__ Ctx) {
    extern __shared__ char smraw[];
    float*          sm = (float*)smraw;                       // [16][SROW] raw scores
    unsigned short* pb = (unsigned short*)(sm + 16 * SROW);   // [16][PROW] bf16 probs
    __shared__ float red[16][17];
    __shared__ float rowmax[16];
    __shared__ float rinv[16];
    __shared__ float ctxacc[4][16][DK + 4];                   // four K-quarter partials

    const int qb   = blockIdx.x & 127;
    const int h    = (blockIdx.x >> 7) & 15;
    const int b    = blockIdx.x >> 11;
    const int wave = threadIdx.x >> 5;
    const int lane = threadIdx.x & 31;

    const unsigned short* Qbase = Qw + ((size_t)b * NH + h) * SEQ * DK;
    const unsigned short* Kbase = Kw + ((size_t)b * NH + h) * SEQ * DK;
    const unsigned short* Vbase = Vt + ((size_t)b * NH + h) * DK * SEQ;

    // Q fragments for dk=64 (two K-steps of 32), reused across all key tiles.
    v16bf aq0 = load_A_b16(Qbase, DK, qb * 16, 0,  lane);
    v16bf aq1 = load_A_b16(Qbase, DK, qb * 16, 32, lane);

    // ---- pass 1: scores (only tiles on/below the diagonal) ----
    for (int kt = wave; kt <= qb; kt += 8) {
        v16bf bk0 = load_B_b16(Kbase, DK, kt * 16, 0,  lane);
        v16bf bk1 = load_B_b16(Kbase, DK, kt * 16, 32, lane);
        v8f c = {};
        c = WMMA_BF16(aq0, bk0, c);
        c = WMMA_BF16(aq1, bk1, c);
#pragma unroll
        for (int i = 0; i < 8; ++i)
            sm[(((lane >> 4) << 3) + i) * SROW + kt * 16 + (lane & 15)] = c[i];
    }
    __syncthreads();

    const int kvPad = (((qb + 1) * 16) + 31) & ~31;   // zero-padded to WMMA K step

    // ---- softmax over each of the 16 query rows; emit bf16 P ----
    {
        const int row = threadIdx.x >> 4, t = threadIdx.x & 15;
        const int qi  = qb * 16 + row;
        float lm = -3.0e38f;
        for (int c = t; c <= qi; c += 16) lm = fmaxf(lm, sm[row * SROW + c]);
        red[row][t] = lm;
        __syncthreads();
        if (t == 0) {
            float m = red[row][0];
            for (int i = 1; i < 16; ++i) m = fmaxf(m, red[row][i]);
            rowmax[row] = m;
        }
        __syncthreads();
        const float mx = rowmax[row];
        float ls = 0.f;
        for (int c = t; c < kvPad; c += 16) {
            float pv = 0.f;
            if (c <= qi) pv = __expf(sm[row * SROW + c] - mx);  // causal mask + pad
            pb[row * PROW + c] = f2bf(pv);
            ls += pv;
        }
        red[row][t] = ls;
        __syncthreads();
        if (t == 0) {
            float s2 = 0.f;
            for (int i = 0; i < 16; ++i) s2 += red[row][i];
            rinv[row] = 1.0f / s2;
        }
    }
    __syncthreads();

    // ---- pass 2: ctx = P @ V ; wave -> (n-tile pair, K quarter) ----
    {
        const int npair = (wave & 1) * 2;     // V d-tiles {0,1} or {2,3}
        const int kq    = wave >> 1;          // K quarter 0..3
        v8f acc0 = {}, acc1 = {};
        for (int k0 = kq * 32; k0 < kvPad; k0 += 128) {
            v16bf aP  = load_A_b16(pb, PROW, 0, k0, lane);                  // ds_load
            v16bf bV0 = load_B_b16(Vbase, SEQ, (npair    ) * 16, k0, lane); // V is [dk,S]
            v16bf bV1 = load_B_b16(Vbase, SEQ, (npair + 1) * 16, k0, lane);
            acc0 = WMMA_BF16(aP, bV0, acc0);
            acc1 = WMMA_BF16(aP, bV1, acc1);
        }
#pragma unroll
        for (int i = 0; i < 8; ++i) {
            int r = ((lane >> 4) << 3) + i;
            ctxacc[kq][r][(npair    ) * 16 + (lane & 15)] = acc0[i];
            ctxacc[kq][r][(npair + 1) * 16 + (lane & 15)] = acc1[i];
        }
    }
    __syncthreads();

    // ---- writeout: sum quarters, 1/rowsum, store bf16 ctx [B,S,H*dk] ----
    for (int e = threadIdx.x; e < 16 * DK; e += 256) {
        int   r = e >> 6, d = e & (DK - 1);
        int   s = qb * 16 + r;
        float v = (ctxacc[0][r][d] + ctxacc[1][r][d] +
                   ctxacc[2][r][d] + ctxacc[3][r][d]) * rinv[r];
        Ctx[((size_t)b * SEQ + s) * D_MODEL + h * DK + d] = f2bf(v);
    }
}

// =====================================================================
// Kernel 3: output projection  out = ctx @ Wo.T + bo  (fp32 out), 4x2 tiles.
// =====================================================================
__global__ void __launch_bounds__(256)
mha_oproj(const unsigned short* __restrict__ Ctx,
          const unsigned short* __restrict__ Wob,
          const float* __restrict__ bo,
          float* __restrict__ out) {
    const int wid  = blockIdx.x * 8 + (threadIdx.x >> 5);
    const int lane = threadIdx.x & 31;
    const int m0   = (wid >> 5) << 6;        // 64 M-blocks of 64 rows
    const int n0   = (wid & 31) << 5;        // 32 N-blocks of 32 cols

    const unsigned short* aRow = Ctx + (size_t)(m0 + (lane & 15)) * D_MODEL;
    const unsigned short* bRow = Wob + (size_t)(n0 + (lane & 15)) * D_MODEL;

    v8f acc[4][2];
#pragma unroll
    for (int tm = 0; tm < 4; ++tm)
#pragma unroll
        for (int tn = 0; tn < 2; ++tn) acc[tm][tn] = (v8f){};

    for (int k = 0; k < D_MODEL; k += 32) {
        if (k + 32 < D_MODEL) {
            __builtin_prefetch(aRow + k + 32, 0, 1);
            __builtin_prefetch(bRow + k + 32, 0, 1);
        }
        v16bf a[4], b[2];
#pragma unroll
        for (int tm = 0; tm < 4; ++tm)
            a[tm] = load_A_b16(Ctx, D_MODEL, m0 + tm * 16, k, lane);
#pragma unroll
        for (int tn = 0; tn < 2; ++tn)
            b[tn] = load_B_b16(Wob, D_MODEL, n0 + tn * 16, k, lane);
#pragma unroll
        for (int tm = 0; tm < 4; ++tm)
#pragma unroll
            for (int tn = 0; tn < 2; ++tn)
                acc[tm][tn] = WMMA_BF16(a[tm], b[tn], acc[tm][tn]);
    }

#pragma unroll
    for (int tm = 0; tm < 4; ++tm) {
#pragma unroll
        for (int tn = 0; tn < 2; ++tn) {
            const int   n    = n0 + tn * 16 + (lane & 15);
            const float bias = bo[n];
#pragma unroll
            for (int i = 0; i < 8; ++i) {
                int m = m0 + tm * 16 + ((lane >> 4) << 3) + i;
                out[(size_t)m * D_MODEL + n] = acc[tm][tn][i] + bias;
            }
        }
    }
}

extern "C" void kernel_launch(void* const* d_in, const int* in_sizes, int n_in,
                              void* d_out, int out_size, void* d_ws, size_t ws_size,
                              hipStream_t stream) {
    (void)in_sizes; (void)n_in; (void)out_size; (void)ws_size;
    const float* x  = (const float*)d_in[0];
    const float* Wq = (const float*)d_in[1];
    const float* bq = (const float*)d_in[2];
    const float* Wk = (const float*)d_in[3];
    const float* bk = (const float*)d_in[4];
    const float* Wv = (const float*)d_in[5];
    const float* bv = (const float*)d_in[6];
    const float* Wo = (const float*)d_in[7];
    const float* bo = (const float*)d_in[8];
    float* out = (float*)d_out;

    const size_t act = (size_t)BATCH * NH * SEQ * DK;   // 4 Mi elements
    const size_t wsz = (size_t)D_MODEL * D_MODEL;       // 1 Mi elements
    unsigned short* Qw  = (unsigned short*)d_ws;
    unsigned short* Kw  = Qw  + act;
    unsigned short* Vt  = Kw  + act;
    unsigned short* Ctx = Vt  + act;
    unsigned short* xb  = Ctx + act;
    unsigned short* Wqb = xb  + act;
    unsigned short* Wkb = Wqb + wsz;
    unsigned short* Wvb = Wkb + wsz;
    unsigned short* Wob = Wvb + wsz;

    // ---- bulk fp32 -> bf16 conversions (x + 4 weight matrices) ----
    mha_cvt_bf16<<<(int)(act / (256 * 8)), 256, 0, stream>>>(x,  xb,  (int)act);
    mha_cvt_bf16<<<(int)(wsz / (256 * 8)), 256, 0, stream>>>(Wq, Wqb, (int)wsz);
    mha_cvt_bf16<<<(int)(wsz / (256 * 8)), 256, 0, stream>>>(Wk, Wkb, (int)wsz);
    mha_cvt_bf16<<<(int)(wsz / (256 * 8)), 256, 0, stream>>>(Wv, Wvb, (int)wsz);
    mha_cvt_bf16<<<(int)(wsz / (256 * 8)), 256, 0, stream>>>(Wo, Wob, (int)wsz);

    // 3 outputs * 64 M-blocks * 32 N-blocks = 6144 wave-blocks / 8 waves
    mha_qkv_proj<<<768, 256, 0, stream>>>(xb, Wqb, Wkb, Wvb, bq, bk, bv, Qw, Kw, Vt);
    // B*H*(S/16) = 4096 blocks; dynamic LDS = f32 scores + bf16 probs
    const size_t dynLds = (size_t)16 * SROW * sizeof(float) + (size_t)16 * PROW * sizeof(unsigned short);
    mha_attention<<<4096, 256, dynLds, stream>>>(Qw, Kw, Vt, Ctx);
    // 64 * 32 = 2048 wave-blocks / 8 waves
    mha_oproj<<<256, 256, 0, stream>>>(Ctx, Wob, bo, out);
}
